// BigBirdSparseAttention_51041391346262
// MI455X (gfx1250) — matrix operand order repaired and checked by reading, not compiled, not run on hardware
//
#include <hip/hip_runtime.h>
#include <hip/hip_bf16.h>

typedef __attribute__((ext_vector_type(16))) _Float16 v16h;
typedef __attribute__((ext_vector_type(8)))  _Float16 v8h;
typedef __attribute__((ext_vector_type(4)))  _Float16 v4h;
typedef __attribute__((ext_vector_type(2)))  _Float16 v2h;
typedef __attribute__((ext_vector_type(8)))  float    v8f;
typedef int v4i_ __attribute__((__vector_size__(16)));

#define EMBED 1024
#define HEADS 16
#define HD    64
#define LSEQ  2048
#define BSZ   2
#define BLK   64
#define NB    32   /* LSEQ / BLK */

__device__ __forceinline__ v16h vcat(v8h lo, v8h hi) {
  v16h r;
#pragma unroll
  for (int i = 0; i < 8; ++i) { r[i] = lo[i]; r[8 + i] = hi[i]; }
  return r;
}

__device__ __forceinline__ v8f wmma_f16(v16h a, v16h b, v8f c) {
  return __builtin_amdgcn_wmma_f32_16x16x32_f16(false, a, false, b, (short)0, c,
                                                false, false);
}

// -------- CDNA5 async global->LDS copy (guarded; falls back to sync) --------
#if defined(__HIP_DEVICE_COMPILE__) && \
    __has_builtin(__builtin_amdgcn_global_load_async_to_lds_b128)
#define HAVE_ASYNC_LDS 1
#else
#define HAVE_ASYNC_LDS 0
#endif

#if HAVE_ASYNC_LDS
__device__ __forceinline__ void async_cp16(const void* g, void* l) {
  // toolchain signature: (int4* src, int4* lds_dst, imm offset, imm cpol)
  __builtin_amdgcn_global_load_async_to_lds_b128((v4i_*)g, (v4i_*)l, 0, 0);
}
__device__ __forceinline__ void async_wait0() {
#if __has_builtin(__builtin_amdgcn_s_wait_asynccnt)
  __builtin_amdgcn_s_wait_asynccnt(0);
#else
  asm volatile("s_wait_asynccnt 0" ::: "memory");
#endif
}
#endif

// ---------------------------------------------------------------------------
// fp32 -> fp16 conversion (vectorized)
// ---------------------------------------------------------------------------
__global__ __launch_bounds__(256) void cvt_f32_f16(const float* __restrict__ s,
                                                   _Float16* __restrict__ d,
                                                   int n) {
  int i = (blockIdx.x * 256 + threadIdx.x) * 4;
  if (i < n) {
    float4 v = *(const float4*)(s + i);
    v4h o;
    o.x = (_Float16)v.x; o.y = (_Float16)v.y;
    o.z = (_Float16)v.z; o.w = (_Float16)v.w;
    *(v4h*)(d + i) = o;
  }
}

// ---------------------------------------------------------------------------
// C[m][n] = sum_k X[m][k] * W[n][k]   (i.e. X @ W^T), f16 in, f16 or f32 out.
// Block = 256 threads = 8 waves; wave tile = 64x64; block tile = 128x256.
// Fragments loaded directly from global: the f16 working set is L2-resident
// (192 MB L2 on MI455X), so cross-wave reuse is absorbed by L2.
// ---------------------------------------------------------------------------
template <bool F32OUT>
__global__ __launch_bounds__(256) void gemm_xwt(const _Float16* __restrict__ X,
                                                const _Float16* __restrict__ W,
                                                _Float16* __restrict__ outh,
                                                float* __restrict__ outf,
                                                int M, int N, int K) {
  int tid  = threadIdx.x;
  int wid  = tid >> 5;
  int lane = tid & 31;
  int hl   = lane >> 4;    // lane half (0/1)
  int lrow = lane & 15;
  int wm = wid >> 2, wn = wid & 3;
  int m0 = blockIdx.x * 128 + wm * 64;
  int n0 = blockIdx.y * 256 + wn * 64;

  v8f acc[4][4] = {};
  for (int k0 = 0; k0 < K; k0 += 32) {
    v16h a[4], bf[4];
#pragma unroll
    for (int i = 0; i < 4; ++i) {
      // A 16x32 frag: lanes 0-15 K 0..7 & 16..23 ; lanes 16-31 K 8..15 & 24..31
      const _Float16* pa = X + (size_t)(m0 + i * 16 + lrow) * K + k0 + hl * 8;
      a[i] = vcat(*(const v8h*)pa, *(const v8h*)(pa + 16));
      // B 32x16 frag: col = W row (n); lanes 0-15 K 0..7 & 16..23 ; lanes 16-31 K 8..15 & 24..31
      const _Float16* pb = W + (size_t)(n0 + i * 16 + lrow) * K + k0 + hl * 8;
      bf[i] = vcat(*(const v8h*)pb, *(const v8h*)(pb + 16));
    }
#pragma unroll
    for (int mi = 0; mi < 4; ++mi)
#pragma unroll
      for (int ni = 0; ni < 4; ++ni)
        acc[mi][ni] = wmma_f16(a[mi], bf[ni], acc[mi][ni]);
  }
#pragma unroll
  for (int mi = 0; mi < 4; ++mi)
#pragma unroll
    for (int ni = 0; ni < 4; ++ni)
#pragma unroll
      for (int i = 0; i < 8; ++i) {
        size_t r = (size_t)(m0 + mi * 16 + i + hl * 8);
        size_t c = (size_t)(n0 + ni * 16 + lrow);
        if constexpr (F32OUT) outf[r * N + c] = acc[mi][ni][i];
        else                  outh[r * N + c] = (_Float16)acc[mi][ni][i];
      }
}

// ---------------------------------------------------------------------------
// Block-sparse attention, one workgroup per (batch, head, q-block).
// Flash-style online softmax over the allowed 64x64 K/V blocks.
// K/V blocks are async-copied global->LDS (ASYNCcnt), fragments then come
// from LDS via aligned ds_load_b128; softmax is 4-way parallel per row.
// ---------------------------------------------------------------------------
__global__ __launch_bounds__(256) void bigbird_attn(
    const _Float16* __restrict__ Qp, const _Float16* __restrict__ Kp,
    const _Float16* __restrict__ Vp, const unsigned char* __restrict__ mask,
    _Float16* __restrict__ Op) {
  __shared__ __align__(16) float    sS[64][66];     // scores
  __shared__ __align__(16) _Float16 sP[64][72];     // exp(S - m)
  __shared__ __align__(16) _Float16 sKr[64 * 64];   // K block, row-major
  __shared__ __align__(16) _Float16 sVr[64 * 64];   // V block, row-major
  __shared__ __align__(16) _Float16 sVt[64][72];    // V block, transposed
  __shared__ float sRmax[4][64], sRsum[4][64];
  __shared__ float sM[64], sL[64], sScale[64];

  int qb = blockIdx.x % NB;
  int h  = (blockIdx.x / NB) % HEADS;
  int b  = blockIdx.x / (NB * HEADS);

  int tid  = threadIdx.x;
  int wid  = tid >> 5;
  int lane = tid & 31;
  int hl   = lane >> 4;
  int lrow = lane & 15;
  int tm   = wid >> 1;     // row subtile 0..3
  int tn2  = wid & 1;      // column half 0..1

  size_t tokbase = (size_t)b * LSEQ;
  size_t colbase = (size_t)h * HD;

  // preload Q fragments (rows tm*16..+15, K = d dimension 0..63)
  v16h qf[2];
#pragma unroll
  for (int ks = 0; ks < 2; ++ks) {
    const _Float16* p = Qp + (tokbase + qb * BLK + tm * 16 + lrow) * EMBED +
                        colbase + ks * 32 + hl * 8;
    qf[ks] = vcat(*(const v8h*)p, *(const v8h*)(p + 16));
  }

  if (tid < 64) { sM[tid] = -1e30f; sL[tid] = 0.f; }
  v8f o[2] = {};
  __syncthreads();

  const unsigned char* mrow = mask + (size_t)(qb * BLK) * LSEQ;
  int r   = tid & 63;      // softmax row
  int seg = tid >> 6;      // softmax column segment (16 cols)

  for (int kb = 0; kb < NB; ++kb) {
    if (!mrow[kb * BLK]) continue;   // uniform across workgroup

    // ---- stage K and V 64x64 blocks into LDS (async copy, 16B per lane) ----
    {
      const char* gk =
          (const char*)(Kp + (tokbase + kb * BLK) * EMBED + colbase);
      const char* gv =
          (const char*)(Vp + (tokbase + kb * BLK) * EMBED + colbase);
      char* lk = (char*)sKr;
      char* lv = (char*)sVr;
#pragma unroll
      for (int hh = 0; hh < 2; ++hh) {
        int oo = tid * 16 + hh * 4096;                  // offset in 8KB tile
        size_t go = (size_t)(oo >> 7) * (EMBED * 2) + (oo & 127);
#if HAVE_ASYNC_LDS
        async_cp16(gk + go, lk + oo);
        async_cp16(gv + go, lv + oo);
#else
        *(v8h*)(lk + oo) = *(const v8h*)(gk + go);
        *(v8h*)(lv + oo) = *(const v8h*)(gv + go);
#endif
      }
#if HAVE_ASYNC_LDS
      async_wait0();
#endif
    }
    __syncthreads();

    // ---- transpose V inside LDS: sVt[d][k] = sVr[k][d] (paired b32 stores) --
    {
      int k2 = (tid & 31) * 2;
      int dg = tid >> 5;                        // 8 d-values per thread
      const _Float16* p0 = sVr + k2 * 64 + dg * 8;
      v8h a0 = *(const v8h*)p0;
      v8h a1 = *(const v8h*)(p0 + 64);
#pragma unroll
      for (int j = 0; j < 8; ++j) {
        v2h pr; pr.x = a0[j]; pr.y = a1[j];
        *(v2h*)&sVt[dg * 8 + j][k2] = pr;
      }
    }

    // ---- S = Q . K^T * 0.125 into LDS (K fragments from LDS) ----
#pragma unroll
    for (int nt = 0; nt < 2; ++nt) {
      int n0 = tn2 * 32 + nt * 16;
      v8f acc = {};
#pragma unroll
      for (int ks = 0; ks < 2; ++ks) {
        const _Float16* p = sKr + (n0 + lrow) * 64 + ks * 32 + hl * 8;
        v16h bf = vcat(*(const v8h*)p, *(const v8h*)(p + 16));
        acc = wmma_f16(qf[ks], bf, acc);
      }
#pragma unroll
      for (int i = 0; i < 8; ++i)
        sS[tm * 16 + i + hl * 8][n0 + lrow] = acc[i] * 0.125f;
    }
    __syncthreads();

    // ---- online softmax, 4 threads per row ----
    {
      float pm = -1e30f;
#pragma unroll
      for (int c = 0; c < 16; ++c) pm = fmaxf(pm, sS[r][seg * 16 + c]);
      sRmax[seg][r] = pm;
    }
    __syncthreads();
    {
      float mb = fmaxf(fmaxf(sRmax[0][r], sRmax[1][r]),
                       fmaxf(sRmax[2][r], sRmax[3][r]));
      float mold = sM[r];
      float mnew = fmaxf(mold, mb);
      float ps = 0.f;
#pragma unroll
      for (int c = 0; c < 16; ++c) {
        float e = __expf(sS[r][seg * 16 + c] - mnew);
        sP[r][seg * 16 + c] = (_Float16)e;
        ps += e;
      }
      sRsum[seg][r] = ps;
      __syncthreads();
      if (seg == 0) {
        float ssum = sRsum[0][r] + sRsum[1][r] + sRsum[2][r] + sRsum[3][r];
        float sc = __expf(mold - mnew);
        sL[r] = sL[r] * sc + ssum;
        sM[r] = mnew;
        sScale[r] = sc;
      }
    }
    __syncthreads();

    // ---- rescale O accumulators, then O += P . V ----
#pragma unroll
    for (int nt = 0; nt < 2; ++nt)
#pragma unroll
      for (int i = 0; i < 8; ++i)
        o[nt][i] *= sScale[tm * 16 + i + hl * 8];

#pragma unroll
    for (int ks = 0; ks < 2; ++ks) {
      const _Float16* pp = &sP[tm * 16 + lrow][ks * 32 + hl * 8];
      v16h pa = vcat(*(const v8h*)pp, *(const v8h*)(pp + 16));
#pragma unroll
      for (int nt = 0; nt < 2; ++nt) {
        int n0 = tn2 * 32 + nt * 16;
        const _Float16* vp = &sVt[n0 + lrow][ks * 32 + hl * 8];
        v16h vb = vcat(*(const v8h*)vp, *(const v8h*)(vp + 16));
        o[nt] = wmma_f16(pa, vb, o[nt]);
      }
    }
    __syncthreads();
  }

  // ---- finalize: divide by row sums, store f16 ----
#pragma unroll
  for (int nt = 0; nt < 2; ++nt) {
    int n0 = tn2 * 32 + nt * 16;
#pragma unroll
    for (int i = 0; i < 8; ++i) {
      int rr = tm * 16 + i + hl * 8;
      float inv = 1.f / sL[rr];
      Op[(tokbase + qb * BLK + rr) * EMBED + colbase + n0 + lrow] =
          (_Float16)(o[nt][i] * inv);
    }
  }
}

// ---------------------------------------------------------------------------
// Host launch
// ---------------------------------------------------------------------------
extern "C" void kernel_launch(void* const* d_in, const int* in_sizes, int n_in,
                              void* d_out, int out_size, void* d_ws,
                              size_t ws_size, hipStream_t stream) {
  const float* X    = (const float*)d_in[0];
  const float* Wq   = (const float*)d_in[1];
  const float* Wk   = (const float*)d_in[2];
  const float* Wv   = (const float*)d_in[3];
  const float* Wo   = (const float*)d_in[4];
  const unsigned char* mask = (const unsigned char*)d_in[5];
  float* out = (float*)d_out;

  const int Mrows = BSZ * LSEQ;                 // 4096
  const size_t NX = (size_t)Mrows * EMBED;      // 4M elements
  const size_t NW = (size_t)EMBED * EMBED;      // 1M elements

  // workspace layout (f16): Xh | Wqh | Wkh | Wvh | Woh | Qh | Kh | Vh | Ah
  _Float16* ws  = (_Float16*)d_ws;
  _Float16* Xh  = ws;
  _Float16* Wqh = Xh + NX;
  _Float16* Wkh = Wqh + NW;
  _Float16* Wvh = Wkh + NW;
  _Float16* Woh = Wvh + NW;
  _Float16* Qh  = Woh + NW;
  _Float16* Kh  = Qh + NX;
  _Float16* Vh  = Kh + NX;
  _Float16* Ah  = Vh + NX;

  auto cvt = [&](const float* s, _Float16* d, size_t n) {
    int blocks = (int)((n / 4 + 255) / 256);
    cvt_f32_f16<<<blocks, 256, 0, stream>>>(s, d, (int)n);
  };
  cvt(X, Xh, NX);
  cvt(Wq, Wqh, NW);
  cvt(Wk, Wkh, NW);
  cvt(Wv, Wvh, NW);
  cvt(Wo, Woh, NW);

  dim3 g(Mrows / 128, EMBED / 256);
  gemm_xwt<false><<<g, 256, 0, stream>>>(Xh, Wqh, Qh, nullptr, Mrows, EMBED, EMBED);
  gemm_xwt<false><<<g, 256, 0, stream>>>(Xh, Wkh, Kh, nullptr, Mrows, EMBED, EMBED);
  gemm_xwt<false><<<g, 256, 0, stream>>>(Xh, Wvh, Vh, nullptr, Mrows, EMBED, EMBED);

  bigbird_attn<<<BSZ * HEADS * NB, 256, 0, stream>>>(Qh, Kh, Vh, mask, Ah);

  gemm_xwt<true><<<g, 256, 0, stream>>>(Ah, Woh, nullptr, out, Mrows, EMBED, EMBED);
}
